// ODEBlock_31576599560299
// MI455X (gfx1250) — matrix operand order repaired and checked by reading, not compiled, not run on hardware
//
#include <hip/hip_runtime.h>

// CNF / neural-ODE integrator for MI455X (gfx1250, wave32, WMMA).
// Fixed-step RK4 (32 steps), forward-mode tangents for the Jacobian trace.
// Hidden-layer matmuls on v_wmma_f32_16x16x32_f16 (f16 in, f32 accumulate).
// Fast paths: v_exp_f32 + v_rcp_f32 tanh, v_sqrt_f32, packed float4 LDS tables.

typedef __attribute__((ext_vector_type(16))) _Float16 v16h;
typedef __attribute__((ext_vector_type(8)))  _Float16 v8h;
typedef __attribute__((ext_vector_type(8)))  float    v8f;

#define HID      64
#define NNUC     3
#define SPB      8              // states per workgroup
#define NWAVE    6              // 2 state-groups x 3 nuclei
#define NTHREADS (NWAVE * 32)
#define NSTEPS   32
#define BATCH    1024

__device__ __forceinline__ float fast_rcp(float x)  { return __builtin_amdgcn_rcpf(x); }
__device__ __forceinline__ float fast_sqrt(float x) { return __builtin_amdgcn_sqrtf(x); }
__device__ __forceinline__ float fast_tanh(float x) {
  // tanh(x) = 1 - 2/(e^{2x}+1); v_exp + v_rcp, ~1e-7 rel err (tol is 1e-3)
  float e = __expf(2.0f * x);
  return 1.0f - 2.0f * fast_rcp(e + 1.0f);
}

__global__ __launch_bounds__(NTHREADS, 1)
void cnf_rk4_wmma(const float* __restrict__ states,
                  const float* __restrict__ W0, const float* __restrict__ b0,
                  const float* __restrict__ W1, const float* __restrict__ b1,
                  const float* __restrict__ W2, const float* __restrict__ b2,
                  const float* __restrict__ W3, const float* __restrict__ b3,
                  const float* __restrict__ nuclei,
                  float* __restrict__ out)
{
  // Wt*[out][in] : transposed weights in f16 so B-layout fragments are contiguous.
  __shared__ _Float16 Wt1[HID][HID];
  __shared__ _Float16 Wt2[HID][HID];
  __shared__ _Float16 act[NWAVE][16][HID];     // per-wave D->A restage buffer
  __shared__ float4 l0c[HID];                  // (W0[0][k], W0[1][k], b0[k], -)
  __shared__ float4 colpk[HID];                // (b1[c], b2[c], W3[c], -)
  __shared__ float nucs[NNUC][3];
  __shared__ float b3s;
  __shared__ float y[SPB][4];                  // integrated state (x, logp)
  __shared__ float xs[SPB][3];                 // current stage position
  __shared__ float ks[SPB][4];                 // RK4 weighted k-sum
  __shared__ float fo[SPB][NNUC][4];           // per-(state,nucleus): a, da/dx[3]

  const int tid = threadIdx.x;

  // ---------------- one-shot preload (HBM -> LDS) ----------------
  for (int i = tid; i < HID * HID; i += NTHREADS) {
    const int in = i >> 6, o = i & 63;         // W[in][out] row-major
    Wt1[o][in] = (_Float16)W1[i];
    Wt2[o][in] = (_Float16)W2[i];
  }
  for (int i = tid; i < HID; i += NTHREADS) {
    l0c[i]   = make_float4(W0[i], W0[HID + i], b0[i], 0.0f);
    colpk[i] = make_float4(b1[i], b2[i], W3[i], 0.0f);
  }
  if (tid < 9)  nucs[tid / 3][tid % 3] = nuclei[tid];
  if (tid == 0) b3s = b3[0];
  for (int i = tid; i < SPB * 4; i += NTHREADS) {
    const int s = i >> 2, j = i & 3;
    const float v = states[(blockIdx.x * SPB + s) * 4 + j];
    y[s][j] = v; ks[s][j] = 0.0f;
    if (j < 3) xs[s][j] = v;
  }
  __syncthreads();

  // ---------------- wave / lane geometry ----------------
  const int lane = tid & 31;
  const int wave = tid >> 5;
  const int grp  = wave / NNUC;                // which 4-state group (0..1)
  const int nuc  = wave % NNUC;                // nucleus handled by this wave
  const int hsel = lane >> 4;                  // lane half (A/B K-half select)
  const int l16  = lane & 15;
  // A-layout: lanes L and L+16 both hold row M = L (different K halves).
  const int sLoc = grp * 4 + (l16 >> 2);       // state index in WG for this row
  const int comp = l16 & 3;                    // 0 = primal, 1..3 = tangent dir

  // ---------------- resident weight fragments (B layout) ----------------
  // B tile (32x16): elem e <-> K = kt*32 + hsel*8 + (e<8 ? e : e+8) (matches A layout).
  v16h B1f[2][4], B2f[2][4];
  #pragma unroll
  for (int kt = 0; kt < 2; ++kt)
    #pragma unroll
    for (int nt = 0; nt < 4; ++nt) {
      const int col = nt * 16 + l16;
      const int kb  = kt * 32 + hsel * 8;
      B1f[kt][nt] = __builtin_shufflevector(*(const v8h*)&Wt1[col][kb], *(const v8h*)&Wt1[col][kb + 16],
          0,1,2,3,4,5,6,7,8,9,10,11,12,13,14,15);
      B2f[kt][nt] = __builtin_shufflevector(*(const v8h*)&Wt2[col][kb], *(const v8h*)&Wt2[col][kb + 16],
          0,1,2,3,4,5,6,7,8,9,10,11,12,13,14,15);
    }

  const v8f zf = {};
  const float dt = 1.0f / NSTEPS;

  for (int step = 0; step < NSTEPS; ++step) {
    const float tbase = step * dt;
    for (int st = 0; st < 4; ++st) {
      const float tcur = tbase + dt * ((st == 0) ? 0.0f : (st == 3) ? 1.0f : 0.5f);

      // ---- per-row geometry (row = (state sLoc, nucleus nuc, comp)) ----
      const float x0 = xs[sLoc][0], x1 = xs[sLoc][1], x2 = xs[sLoc][2];
      const float z0 = x0 - nucs[nuc][0];
      const float z1 = x1 - nucs[nuc][1];
      const float z2 = x2 - nucs[nuc][2];
      const float r  = fast_sqrt(z0 * z0 + z1 * z1 + z2 * z2);
      const float zc = (comp == 1) ? z0 : ((comp == 2) ? z1 : z2);
      const float drdx = zc * fast_rcp(fmaxf(r, 1e-12f)); // dr/dx_comp

      // ---- layer 0 (2->64) computed per lane, emitted directly in A layout ----
      // A-layout: elem e <-> K = kt*32 + hsel*8 + (e<8 ? e : e+8)
      v16h A0, A1;
      #pragma unroll
      for (int e = 0; e < 16; ++e) {
        const int koff = hsel * 8 + ((e < 8) ? e : (e + 8));
        {
          const float4 c0 = l0c[koff];
          const float hp = fast_tanh(tcur * c0.x + r * c0.y + c0.z);
          const float v  = (comp == 0) ? hp : (1.0f - hp * hp) * (drdx * c0.y);
          A0[e] = (_Float16)v;
        }
        {
          const float4 c1 = l0c[32 + koff];
          const float hp = fast_tanh(tcur * c1.x + r * c1.y + c1.z);
          const float v  = (comp == 0) ? hp : (1.0f - hp * hp) * (drdx * c1.y);
          A1[e] = (_Float16)v;
        }
      }

      // ---- layer 1: (16x64) @ (64x64) via 8 WMMAs ----
      v8f acc[4];
      #pragma unroll
      for (int nt = 0; nt < 4; ++nt) {
        v8f c = __builtin_amdgcn_wmma_f32_16x16x32_f16(
            false, A0, false, B1f[0][nt], (short)0, zf, false, false);
        acc[nt] = __builtin_amdgcn_wmma_f32_16x16x32_f16(
            false, A1, false, B1f[1][nt], (short)0, c, false, false);
      }

      // per-column constants (b1, b2, w3) fetched once, reused across layers
      float4 cps[4];
      #pragma unroll
      for (int nt = 0; nt < 4; ++nt) cps[nt] = colpk[nt * 16 + l16];

      // ---- tanh / tanh' in D layout (primal row is lane-local: VGPR v & ~3) ----
      #pragma unroll
      for (int nt = 0; nt < 4; ++nt) {
        const int col = nt * 16 + l16;
        const float bb = cps[nt].x;
        const float h0 = fast_tanh(acc[nt][0] + bb);
        const float h4 = fast_tanh(acc[nt][4] + bb);
        const float g0 = 1.0f - h0 * h0, g4 = 1.0f - h4 * h4;
        const int r0 = 8 * hsel;                 // rows v / v+8 per lane half
        act[wave][r0 + 0][col] = (_Float16)h0;
        act[wave][r0 + 1][col] = (_Float16)(g0 * acc[nt][1]);
        act[wave][r0 + 2][col] = (_Float16)(g0 * acc[nt][2]);
        act[wave][r0 + 3][col] = (_Float16)(g0 * acc[nt][3]);
        act[wave][r0 + 4][col] = (_Float16)h4;
        act[wave][r0 + 5][col] = (_Float16)(g4 * acc[nt][5]);
        act[wave][r0 + 6][col] = (_Float16)(g4 * acc[nt][6]);
        act[wave][r0 + 7][col] = (_Float16)(g4 * acc[nt][7]);
      }

      // ---- reload in A layout (per-wave private buffer, wave-ordered LDS) ----
      const int kb0 = hsel * 8;
      const v8h lo0 = *(const v8h*)&act[wave][l16][kb0];
      const v8h hi0 = *(const v8h*)&act[wave][l16][kb0 + 16];
      const v8h lo1 = *(const v8h*)&act[wave][l16][kb0 + 32];
      const v8h hi1 = *(const v8h*)&act[wave][l16][kb0 + 48];
      const v16h A20 = __builtin_shufflevector(lo0, hi0,
          0,1,2,3,4,5,6,7,8,9,10,11,12,13,14,15);
      const v16h A21 = __builtin_shufflevector(lo1, hi1,
          0,1,2,3,4,5,6,7,8,9,10,11,12,13,14,15);

      // ---- layer 2: 8 more WMMAs ----
      v8f acc2[4];
      #pragma unroll
      for (int nt = 0; nt < 4; ++nt) {
        v8f c = __builtin_amdgcn_wmma_f32_16x16x32_f16(
            false, A20, false, B2f[0][nt], (short)0, zf, false, false);
        acc2[nt] = __builtin_amdgcn_wmma_f32_16x16x32_f16(
            false, A21, false, B2f[1][nt], (short)0, c, false, false);
      }

      // ---- layer 2 nonlinearity fused with layer-3 (64->1) dot ----
      float p[8] = {0, 0, 0, 0, 0, 0, 0, 0};
      #pragma unroll
      for (int nt = 0; nt < 4; ++nt) {
        const float bb = cps[nt].y;
        const float wc = cps[nt].z;
        const float h0 = fast_tanh(acc2[nt][0] + bb);
        const float h4 = fast_tanh(acc2[nt][4] + bb);
        const float g0 = 1.0f - h0 * h0, g4 = 1.0f - h4 * h4;
        p[0] += h0 * wc;
        p[1] += g0 * acc2[nt][1] * wc;
        p[2] += g0 * acc2[nt][2] * wc;
        p[3] += g0 * acc2[nt][3] * wc;
        p[4] += h4 * wc;
        p[5] += g4 * acc2[nt][5] * wc;
        p[6] += g4 * acc2[nt][6] * wc;
        p[7] += g4 * acc2[nt][7] * wc;
      }
      // reduce over the 16 lanes of each half (rows 0-7 live in lanes 0-15)
      #pragma unroll
      for (int off = 1; off < 16; off <<= 1) {
        #pragma unroll
        for (int v = 0; v < 8; ++v) p[v] += __shfl_xor(p[v], off, 32);
      }
      if (l16 == 0) {
        #pragma unroll
        for (int v = 0; v < 8; ++v) {
          const int row = v + 8 * hsel;
          const int s = grp * 4 + (row >> 2);
          const int c = row & 3;
          fo[s][nuc][c] = p[v] + ((c == 0) ? b3s : 0.0f);
        }
      }
      __syncthreads();

      // ---- nucleus-sum + RK4 bookkeeping (one thread per state) ----
      if (tid < SPB) {
        const int s = tid;
        float dz0 = 0, dz1 = 0, dz2 = 0, tr = 0;
        #pragma unroll
        for (int n = 0; n < NNUC; ++n) {
          const float a  = fo[s][n][0];
          const float zx = xs[s][0] - nucs[n][0];
          const float zy = xs[s][1] - nucs[n][1];
          const float zz = xs[s][2] - nucs[n][2];
          dz0 += a * zx; dz1 += a * zy; dz2 += a * zz;
          tr  += fo[s][n][1] * zx + fo[s][n][2] * zy + fo[s][n][3] * zz + 3.0f * a;
        }
        const float f0 = dz0, f1 = dz1, f2 = dz2, f3 = -tr;   // T0 = 1
        const float w = (st == 0 || st == 3) ? 1.0f : 2.0f;
        ks[s][0] += w * f0; ks[s][1] += w * f1;
        ks[s][2] += w * f2; ks[s][3] += w * f3;
        if (st < 3) {
          const float al = (st == 2) ? 1.0f : 0.5f;
          xs[s][0] = y[s][0] + dt * al * f0;
          xs[s][1] = y[s][1] + dt * al * f1;
          xs[s][2] = y[s][2] + dt * al * f2;
        } else {
          const float c6 = dt / 6.0f;
          y[s][0] += c6 * ks[s][0]; y[s][1] += c6 * ks[s][1];
          y[s][2] += c6 * ks[s][2]; y[s][3] += c6 * ks[s][3];
          ks[s][0] = ks[s][1] = ks[s][2] = ks[s][3] = 0.0f;
          xs[s][0] = y[s][0]; xs[s][1] = y[s][1]; xs[s][2] = y[s][2];
        }
      }
      __syncthreads();
    }
  }

  if (tid < SPB * 4) {
    const int s = tid >> 2, j = tid & 3;
    out[(blockIdx.x * SPB + s) * 4 + j] = y[s][j];
  }
}

extern "C" void kernel_launch(void* const* d_in, const int* in_sizes, int n_in,
                              void* d_out, int out_size, void* d_ws, size_t ws_size,
                              hipStream_t stream)
{
  (void)in_sizes; (void)n_in; (void)d_ws; (void)ws_size; (void)out_size;
  const float* states = (const float*)d_in[0];
  const float* W0 = (const float*)d_in[1];
  const float* b0 = (const float*)d_in[2];
  const float* W1 = (const float*)d_in[3];
  const float* b1 = (const float*)d_in[4];
  const float* W2 = (const float*)d_in[5];
  const float* b2 = (const float*)d_in[6];
  const float* W3 = (const float*)d_in[7];
  const float* b3 = (const float*)d_in[8];
  const float* nuclei = (const float*)d_in[9];

  dim3 grid(BATCH / SPB);
  dim3 block(NTHREADS);
  cnf_rk4_wmma<<<grid, block, 0, stream>>>(states, W0, b0, W1, b1, W2, b2,
                                           W3, b3, nuclei, (float*)d_out);
}
